// AdaBlock_62070867362353
// MI455X (gfx1250) — matrix operand order repaired and checked by reading, not compiled, not run on hardware
//
#include <hip/hip_runtime.h>
#include <hip/hip_bf16.h>
#include <math.h>

// ---------------------------------------------------------------------------
// CDNA5 (gfx1250) AdaBlock: dwconv7x7 -> route -> {LN+MLP(GELU), LN+Linear}
// -> scatter -> residual.  GEMMs via v_wmma_f32_16x16x32_bf16 (wave32).
// ---------------------------------------------------------------------------

typedef __attribute__((ext_vector_type(16))) __bf16 v16bf;
typedef __attribute__((ext_vector_type(8)))  float  v8f;

#define Cdim 256
#define Hdim 128
#define Wdim 128
#define HWdim 16384
#define Ndim 4

__device__ __forceinline__ unsigned short f2bf(float f) {
    unsigned int u = __float_as_uint(f);
    unsigned int r = u + 0x7FFFu + ((u >> 16) & 1u);   // round-to-nearest-even
    return (unsigned short)(r >> 16);
}

union Frag {         // one 16x32 bf16 WMMA operand fragment (per-lane view)
    v16bf v;
    uint4 q[2];
};

// ---------------------------------------------------------------------------
// 1) NCHW -> NHWC transpose (x -> xt), LDS tiled 32x32.
//    Tile staged via CDNA5 async global->LDS copy (ASYNCcnt).
// ---------------------------------------------------------------------------
__global__ __launch_bounds__(256)
void k_t_in(const float* __restrict__ X, float* __restrict__ Xt) {
    __shared__ float tile[32][33];
    int n = blockIdx.z;
    int hw0 = blockIdx.x * 32, c0 = blockIdx.y * 32;
    int tx = threadIdx.x & 31, ty = threadIdx.x >> 5;       // ty: 0..7
    const float* src = X + (size_t)n * Cdim * HWdim;
#pragma unroll
    for (int i = 0; i < 4; ++i) {
        int c = c0 + ty + i * 8;
        unsigned lds_off = (unsigned)(size_t)&tile[ty + i * 8][tx];
        const float* g = src + (size_t)c * HWdim + hw0 + tx;
        asm volatile("global_load_async_to_lds_b32 %0, %1, off"
                     :: "v"(lds_off), "v"(g) : "memory");
    }
    asm volatile("s_wait_asynccnt 0x0" ::: "memory");
    __syncthreads();
    float* dst = Xt + (size_t)n * HWdim * Cdim;
#pragma unroll
    for (int i = 0; i < 4; ++i) {
        int hw = hw0 + ty + i * 8;
        dst[(size_t)hw * Cdim + c0 + tx] = tile[tx][ty + i * 8];
    }
}

// ---------------------------------------------------------------------------
// 2) Depthwise 7x7 conv in NHWC, pad 3.  One thread per (n,h,c) and TWO w
//    pixels, sharing the 7x8 input window (halves loads).  Coalesced along c.
// ---------------------------------------------------------------------------
__global__ __launch_bounds__(256)
void k_dwconv(const float* __restrict__ Xt, const float* __restrict__ Wd,
              const float* __restrict__ Bd, float* __restrict__ Y) {
    int c = threadIdx.x;
    int w0 = blockIdx.x * 2;
    int h = blockIdx.y, n = blockIdx.z;
    float acc0 = Bd[c];
    float acc1 = acc0;
    const float* wd = Wd + c * 49;
    const float* xb = Xt + (size_t)n * Hdim * Wdim * Cdim + c;
#pragma unroll
    for (int dh = -3; dh <= 3; ++dh) {
        int hh = h + dh;
        if ((unsigned)hh >= (unsigned)Hdim) continue;
        const float* xr = xb + (size_t)hh * Wdim * Cdim;
        const float* wr = wd + (dh + 3) * 7;
#pragma unroll
        for (int k = 0; k < 8; ++k) {
            int ww = w0 - 3 + k;
            float xv = ((unsigned)ww < (unsigned)Wdim)
                           ? xr[(size_t)ww * Cdim] : 0.f;
            if (k < 7) acc0 = fmaf(xv, wr[k], acc0);
            if (k > 0) acc1 = fmaf(xv, wr[k - 1], acc1);
        }
    }
    float* yb = Y + (((size_t)n * Hdim + h) * Wdim + w0) * Cdim + c;
    yb[0] = acc0;
    yb[Cdim] = acc1;
}

// ---------------------------------------------------------------------------
// 3) Gather routed tokens + LayerNorm + cast to bf16 rows (WMMA A source).
//    One wave32 per token; 8 f32 per lane (two float4); shuffle reduction.
// ---------------------------------------------------------------------------
__global__ __launch_bounds__(256)
void k_gather_ln(const float* __restrict__ Y, const long long* __restrict__ idx,
                 const float* __restrict__ g, const float* __restrict__ b,
                 unsigned short* __restrict__ A, int Lr) {
    int lane = threadIdx.x & 31;
    int wave = threadIdx.x >> 5;
    int t = blockIdx.x * 8 + wave;            // token id in [0, N*Lr)
    int n = t / Lr, l = t - n * Lr;
    long long row = idx[(size_t)n * Lr + l];
    const float* src = Y + (((size_t)n * HWdim) + (size_t)row) * Cdim + lane * 8;
    float4 xa = *(const float4*)(src);
    float4 xb = *(const float4*)(src + 4);
    float x[8] = {xa.x, xa.y, xa.z, xa.w, xb.x, xb.y, xb.z, xb.w};
    float s = 0.f, s2 = 0.f;
#pragma unroll
    for (int i = 0; i < 8; ++i) { s += x[i]; s2 += x[i] * x[i]; }
#pragma unroll
    for (int off = 16; off > 0; off >>= 1) {
        s  += __shfl_xor(s,  off);
        s2 += __shfl_xor(s2, off);
    }
    float mean = s * (1.0f / 256.0f);
    float var  = s2 * (1.0f / 256.0f) - mean * mean;
    float inv  = rsqrtf(var + 1e-6f);
    unsigned short* dst = A + (size_t)t * Cdim + lane * 8;
#pragma unroll
    for (int i = 0; i < 8; ++i) {
        int c = lane * 8 + i;
        dst[i] = f2bf((x[i] - mean) * inv * g[c] + b[c]);
    }
}

// ---------------------------------------------------------------------------
// 4) Pack a (K x Nc) f32 row-major weight into WMMA B fragments (bf16).
//    Fragment (kt,nt): 32 lanes x 16 bf16; lane holds column n0+lane%16,
//    K range [kt*32 + (lane/16)*16, +16) contiguous -> two b128 loads in GEMM.
// ---------------------------------------------------------------------------
__global__ __launch_bounds__(256)
void k_pack_b(const float* __restrict__ W, unsigned short* __restrict__ Bp,
              int K, int Nc) {
    int t = blockIdx.x * blockDim.x + threadIdx.x;
    int lane = t & 31;
    int f = t >> 5;
    int nfn = Nc >> 4;
    int kt = f / nfn, nt = f - kt * nfn;
    if (kt >= (K >> 5)) return;
    int n = nt * 16 + (lane & 15);
    int half = lane >> 4;
    int kb = kt * 32 + half * 16;
    unsigned short* dst = Bp + (size_t)f * 512 + lane * 16;
#pragma unroll
    for (int j = 0; j < 16; ++j)
        dst[j] = f2bf(W[(size_t)(kb + j) * Nc + n]);
}

// ---------------------------------------------------------------------------
// 5) WMMA GEMM: D = A(bf16, MxK row-major) x Bp(packed) + bias, epilogue:
//      ep==0 : exact GELU, store bf16 row-major (ld = Nc)
//      ep==1 : val = gamma[col]*val; scatter rows via idx into Ztok (N,HW,C)
//    Block: 8 waves (4 along M x 2 along N); wave tile 64x64 (16 wmma /
//    16 b128-loads per K-step); block tile 256x128.  A & B stream from L2
//    via pointer-bumped b128 loads.
// ---------------------------------------------------------------------------
__global__ __launch_bounds__(256)
void k_gemm_wmma(const unsigned short* __restrict__ A,
                 const unsigned short* __restrict__ Bp,
                 const float* __restrict__ bias,
                 const float* __restrict__ gamma,
                 unsigned short* __restrict__ OutBf,
                 const long long* __restrict__ idx,
                 float* __restrict__ Ztok,
                 int M, int K, int Nc, int ep, int Lr) {
    int lane = threadIdx.x & 31;
    int wave = threadIdx.x >> 5;
    int wm = wave & 3, wn = wave >> 2;          // 4 M-waves x 2 N-waves
    int m0 = blockIdx.y * 256 + wm * 64;
    int n0 = blockIdx.x * 128 + wn * 64;
    int r = lane & 15, half = lane >> 4;

    v8f acc[4][4];
#pragma unroll
    for (int i = 0; i < 4; ++i)
#pragma unroll
        for (int j = 0; j < 4; ++j) acc[i][j] = {};

    // Per-lane streaming pointers (bumped by constant strides in the loop).
    const unsigned short* pa[4];
#pragma unroll
    for (int i = 0; i < 4; ++i)
        pa[i] = A + (size_t)(m0 + i * 16 + r) * K + half * 16;
    int nfn = Nc >> 4;
    const unsigned short* pb[4];
#pragma unroll
    for (int j = 0; j < 4; ++j)
        pb[j] = Bp + (size_t)((n0 + j * 16) >> 4) * 512 + lane * 16;
    const size_t bstride = (size_t)nfn * 512;

    int ksteps = K >> 5;
    for (int kt = 0; kt < ksteps; ++kt) {
        Frag a[4], bfr[4];
#pragma unroll
        for (int i = 0; i < 4; ++i) {
            a[i].q[0] = *(const uint4*)(pa[i]);
            a[i].q[1] = *(const uint4*)(pa[i] + 8);
            if (kt + 1 < ksteps) __builtin_prefetch(pa[i] + 32, 0, 3);
            pa[i] += 32;
        }
#pragma unroll
        for (int j = 0; j < 4; ++j) {
            bfr[j].q[0] = *(const uint4*)(pb[j]);
            bfr[j].q[1] = *(const uint4*)(pb[j] + 8);
            pb[j] += bstride;
        }
#pragma unroll
        for (int i = 0; i < 4; ++i)
#pragma unroll
            for (int j = 0; j < 4; ++j)
                acc[i][j] = __builtin_amdgcn_wmma_f32_16x16x32_bf16(
                    false, a[i].v, false, bfr[j].v, (short)0, acc[i][j],
                    false, false);
    }

    // Epilogue.  C/D layout: lanes 0-15 -> M=v, N=lane; lanes 16-31 -> M=8+v.
    if (ep == 0) {
#pragma unroll
        for (int i = 0; i < 4; ++i) {
#pragma unroll
            for (int j = 0; j < 4; ++j) {
                int col = n0 + j * 16 + r;
                float bb = bias[col];
#pragma unroll
                for (int v = 0; v < 8; ++v) {
                    int m = m0 + i * 16 + half * 8 + v;
                    float val = acc[i][j][v] + bb;
                    float gelu = 0.5f * val * (1.0f + erff(val * 0.70710678118f));
                    OutBf[(size_t)m * Nc + col] = f2bf(gelu);
                }
            }
        }
    } else {
        // Whole 256-row block tile lies in one batch (Lr % 256 == 0).
        int nb = m0 / Lr;
        const long long* idxb = idx + (size_t)nb * Lr + (m0 - nb * Lr);
        float* zb = Ztok + (size_t)nb * HWdim * Cdim;
#pragma unroll
        for (int i = 0; i < 4; ++i) {
            long long rows[8];
#pragma unroll
            for (int v = 0; v < 8; ++v)
                rows[v] = idxb[i * 16 + half * 8 + v];
#pragma unroll
            for (int j = 0; j < 4; ++j) {
                int col = n0 + j * 16 + r;
                float bb = bias[col];
                float gg = gamma[col];
#pragma unroll
                for (int v = 0; v < 8; ++v)
                    zb[(size_t)rows[v] * Cdim + col] = gg * (acc[i][j][v] + bb);
            }
        }
    }
}

// ---------------------------------------------------------------------------
// 6) out = x + transpose(z): read z (N,HW,C) coalesced, LDS transpose,
//    add x (N,C,H,W) coalesced, write out (N,C,H,W).
// ---------------------------------------------------------------------------
__global__ __launch_bounds__(256)
void k_final_add(const float* __restrict__ X, const float* __restrict__ Z,
                 float* __restrict__ Outp) {
    __shared__ float tile[32][33];
    int n = blockIdx.z;
    int hw0 = blockIdx.x * 32, c0 = blockIdx.y * 32;
    int tx = threadIdx.x & 31, ty = threadIdx.x >> 5;
    const float* zsrc = Z + (size_t)n * HWdim * Cdim;
#pragma unroll
    for (int i = 0; i < 4; ++i) {
        int hw = hw0 + ty + i * 8;
        tile[ty + i * 8][tx] = zsrc[(size_t)hw * Cdim + c0 + tx];
    }
    __syncthreads();
    const float* xsrc = X + (size_t)n * Cdim * HWdim;
    float* dst = Outp + (size_t)n * Cdim * HWdim;
#pragma unroll
    for (int i = 0; i < 4; ++i) {
        int c = c0 + ty + i * 8;
        size_t o = (size_t)c * HWdim + hw0 + tx;
        dst[o] = xsrc[o] + tile[tx][ty + i * 8];
    }
}

// ---------------------------------------------------------------------------
extern "C" void kernel_launch(void* const* d_in, const int* in_sizes, int n_in,
                              void* d_out, int out_size, void* d_ws, size_t ws_size,
                              hipStream_t stream) {
    const float* x      = (const float*)d_in[0];
    const float* dw_w   = (const float*)d_in[1];
    const float* dw_b   = (const float*)d_in[2];
    const float* ln_g   = (const float*)d_in[3];
    const float* ln_b   = (const float*)d_in[4];
    const float* w1     = (const float*)d_in[5];
    const float* b1     = (const float*)d_in[6];
    const float* w2     = (const float*)d_in[7];
    const float* b2     = (const float*)d_in[8];
    const float* gamma  = (const float*)d_in[9];
    const float* fp_ln_g= (const float*)d_in[10];
    const float* fp_ln_b= (const float*)d_in[11];
    const float* fp_w   = (const float*)d_in[12];
    const float* fp_b   = (const float*)d_in[13];
    const float* fp_gam = (const float*)d_in[14];
    const long long* idx1 = (const long long*)d_in[15];
    const long long* idx2 = (const long long*)d_in[16];
    float* out = (float*)d_out;

    const int Lr = HWdim / 2;            // 8192
    const int Mtot = Ndim * Lr;          // 32768 routed tokens per set
    const size_t MB = 1u << 20;

    char* ws = (char*)d_ws;
    float*          xt   = (float*)(ws);                 // 64MB (reused as z)
    float*          ytok = (float*)(ws + 64 * MB);       // 64MB (reused as H)
    unsigned short* A1   = (unsigned short*)(ws + 128 * MB); // 16MB
    unsigned short* A2   = (unsigned short*)(ws + 144 * MB); // 16MB
    unsigned short* Hb   = (unsigned short*)(ws + 64 * MB);  // 64MB (over ytok)
    float*          ztok = xt;                               // 64MB (over xt)
    unsigned short* w1p  = (unsigned short*)(ws + 160 * MB); // 512KB
    unsigned short* w2p  = (unsigned short*)(ws + 160 * MB + 512 * 1024);
    unsigned short* fpwp = (unsigned short*)(ws + 161 * MB); // 128KB

    // Weight packing (tiny, once per call)
    k_pack_b<<<dim3((8 * 64 * 32 + 255) / 256), 256, 0, stream>>>(w1, w1p, 256, 1024);
    k_pack_b<<<dim3((32 * 16 * 32 + 255) / 256), 256, 0, stream>>>(w2, w2p, 1024, 256);
    k_pack_b<<<dim3((8 * 16 * 32 + 255) / 256), 256, 0, stream>>>(fp_w, fpwp, 256, 256);

    // x -> NHWC
    k_t_in<<<dim3(HWdim / 32, Cdim / 32, Ndim), 256, 0, stream>>>(x, xt);
    // depthwise conv (2 pixels per thread)
    k_dwconv<<<dim3(Wdim / 2, Hdim, Ndim), 256, 0, stream>>>(xt, dw_w, dw_b, ytok);
    // gather + LN -> bf16 A matrices
    k_gather_ln<<<dim3(Mtot / 8), 256, 0, stream>>>(ytok, idx1, ln_g, ln_b, A1, Lr);
    k_gather_ln<<<dim3(Mtot / 8), 256, 0, stream>>>(ytok, idx2, fp_ln_g, fp_ln_b, A2, Lr);

    // GEMM1: H = gelu(A1 @ w1 + b1)   (32768 x 1024, K=256)
    k_gemm_wmma<<<dim3(1024 / 128, Mtot / 256), 256, 0, stream>>>(
        A1, w1p, b1, nullptr, Hb, idx1, ztok, Mtot, 256, 1024, 0, Lr);
    // GEMM2: z[idx1] = gamma * (H @ w2 + b2)   (32768 x 256, K=1024)
    k_gemm_wmma<<<dim3(256 / 128, Mtot / 256), 256, 0, stream>>>(
        Hb, w2p, b2, gamma, nullptr, idx1, ztok, Mtot, 1024, 256, 1, Lr);
    // Fast path: z[idx2] = fp_gamma * (A2 @ fp_w + fp_b)  (32768 x 256, K=256)
    k_gemm_wmma<<<dim3(256 / 128, Mtot / 256), 256, 0, stream>>>(
        A2, fpwp, fp_b, fp_gam, nullptr, idx2, ztok, Mtot, 256, 256, 1, Lr);

    // out = x + z^T
    k_final_add<<<dim3(HWdim / 32, Cdim / 32, Ndim), 256, 0, stream>>>(x, ztok, out);
}
